// WinEncoderTransformer_70514773066140
// MI455X (gfx1250) — matrix operand order repaired and checked by reading, not compiled, not run on hardware
//
#include <hip/hip_runtime.h>
#include <hip/hip_bf16.h>

typedef __attribute__((ext_vector_type(16))) _Float16 v16h;
typedef __attribute__((ext_vector_type(8)))  float    v8f;

#define D_MODEL 256
#define NHEAD   8
#define DH      32
#define D_FF    512
#define BB      4
#define HH_     128
#define WW_     128
#define T_TOK   (BB * HH_ * WW_)                 // 65536 tokens per layer
#define QKV_STRIDE ((size_t)T_TOK * D_MODEL)     // elems between Q,K,V planes

struct LayerP {
  int s, wh, ww, shH, shW, useMask, L, log2L, NW, nwW, newH, newW;
};

// ---------------- WMMA fragment helpers (CDNA5 16x16x32 f16 layouts) --------
// A (16x32, MxK): lane 0-15 -> M=lane, halves 0..7 = K 0..7, halves 8..15 = K 16..23
//                 lane 16-31 -> M=lane-16, halves = K 8..15 / K 24..31
__device__ inline v16h ld_frag_a(const _Float16* tile, int stride) {
  int lane = threadIdx.x & 31;
  const _Float16* p = tile + (lane & 15) * stride + ((lane >> 4) << 3);
  v16h r;
  float4* rp = reinterpret_cast<float4*>(&r);
  rp[0] = *reinterpret_cast<const float4*>(p);
  rp[1] = *reinterpret_cast<const float4*>(p + 16);
  return r;
}
// B (32x16, KxN), stored as [N][K] rows: lane 0-15 -> N=lane, K 0..7 / K 16..23;
//                 lane 16-31 -> N=lane-16, K 8..15 / K 24..31
__device__ inline v16h ld_frag_b(const _Float16* tile, int stride) {
  int lane = threadIdx.x & 31;
  const _Float16* p = tile + (lane & 15) * stride + ((lane >> 4) << 3);
  v16h r;
  float4* rp = reinterpret_cast<float4*>(&r);
  rp[0] = *reinterpret_cast<const float4*>(p);
  rp[1] = *reinterpret_cast<const float4*>(p + 16);
  return r;
}
__device__ inline v8f wmma16(v16h a, v16h b, v8f c) {
  return __builtin_amdgcn_wmma_f32_16x16x32_f16(false, a, false, b, (short)0, c,
                                                false, false);
}

// ---------------- gfx1250 async global->LDS copy (inline asm, ASYNCcnt) ----
// Generic shared pointer low 32 bits are the wave-relative LDS address
// (flat addressing truncation, ISA 10.2).
__device__ inline unsigned lds_addr32(const void* p) {
  return (unsigned)(unsigned long long)p;
}
__device__ inline void async_copy_b128(void* lds_dst, const _Float16* gsrc) {
  unsigned d = lds_addr32(lds_dst);
  asm volatile("global_load_async_to_lds_b128 %0, %1, off"
               :: "v"(d), "v"(gsrc)
               : "memory");
}
__device__ inline void wait_async_le6() {
  asm volatile("s_wait_asynccnt 0x6" ::: "memory");
}
__device__ inline void wait_async_le0() {
  asm volatile("s_wait_asynccnt 0x0" ::: "memory");
}

// ---------------- weight fp32 -> f16 ----------------------------------------
__global__ __launch_bounds__(256) void cvt_kernel(const float* __restrict__ in,
                                                  _Float16* __restrict__ out, int n) {
  int i = blockIdx.x * 256 + threadIdx.x;
  if (i < n) out[i] = (_Float16)in[i];
}

// ---------------- gather: NCHW -> token-major (roll + dilate + window) -------
__global__ __launch_bounds__(256) void gather_kernel(
    const float* __restrict__ src, const float* __restrict__ pos,
    const unsigned char* __restrict__ mask,
    _Float16* __restrict__ X, _Float16* __restrict__ XP,
    unsigned char* __restrict__ REG, LayerP lp) {
  int idx = blockIdx.x * 256 + threadIdx.x;        // over B*C*H*W (rolled coords)
  int wr = idx & (WW_ - 1);
  int hr = (idx >> 7) & (HH_ - 1);
  int c  = (idx >> 14) & (D_MODEL - 1);
  int b  = idx >> 22;
  int hs = (hr + lp.shH) & (HH_ - 1);              // source coords (roll)
  int ws = (wr + lp.shW) & (WW_ - 1);
  size_t gsrc = (((size_t)b * D_MODEL + c) * HH_ + hs) * WW_ + ws;
  float v  = src[gsrc];
  float pv = pos[gsrc];
  int si = hr % lp.s, hh = hr / lp.s;              // dilate_fold
  int sj = wr % lp.s, wf = wr / lp.s;
  int beff = (b * lp.s + si) * lp.s + sj;
  int wi = hh / lp.wh, ph = hh % lp.wh;            // window_partition
  int wj = wf / lp.ww, pw = wf % lp.ww;
  int nw = wi * lp.nwW + wj;
  int l  = ph * lp.ww + pw;
  size_t t = (size_t)(beff * lp.NW + nw) * lp.L + l;
  X [t * D_MODEL + c] = (_Float16)v;
  XP[t * D_MODEL + c] = (_Float16)(v + pv);
  if (c == 0) {
    int rh = (hh < lp.newH - lp.wh) ? 0 : (hh < lp.newH - lp.wh / 2 ? 1 : 2);
    int rw = (wf < lp.newW - lp.ww) ? 0 : (wf < lp.newW - lp.ww / 2 ? 1 : 2);
    unsigned char reg = lp.useMask ? (unsigned char)(rh * 3 + rw) : (unsigned char)0;
    if (mask[((size_t)b * HH_ + hs) * WW_ + ws]) reg |= 0x80;   // key padding
    REG[t] = reg;
  }
}

// ---------------- tiled WMMA GEMM: Y[M,N] = A[M,K] * W[N,K]^T + bias ---------
// Double-buffered LDS, async global->LDS staging (ASYNCcnt), K-step 64.
// MODE 0: QKV epilogue (A2 = X for the V columns), f16 scatter to per-head layout
// MODE 1: f32 store     MODE 2: ReLU + f16 store
template <int MODE>
__global__ __launch_bounds__(256) void gemm_kernel(
    const _Float16* __restrict__ A, const _Float16* __restrict__ A2,
    const _Float16* __restrict__ Wt, const float* __restrict__ bias,
    void* __restrict__ outp, int K, int N, int log2L) {
  constexpr int KS = 64;                 // K per step
  constexpr int LDA = 72;                // padded LDS row (halves)
  __shared__ __align__(16) _Float16 sA[2][128 * LDA];
  __shared__ __align__(16) _Float16 sB[2][64 * LDA];
  int tid = threadIdx.x;
  int wave = tid >> 5, lane = tid & 31;
  int mBase = blockIdx.x * 128;
  int nBase = blockIdx.y * 64;
  const _Float16* Asrc = (MODE == 0 && nBase >= 512) ? A2 : A;
  int wm = (wave >> 1) * 32, wn = (wave & 1) * 32;
  v8f z = {0.f, 0.f, 0.f, 0.f, 0.f, 0.f, 0.f, 0.f};
  v8f acc00 = z, acc01 = z, acc10 = z, acc11 = z;

  auto issue_tile = [&](int buf, int ks) {
    // A tile: 128 rows x 64 halves = 1024 x 16B; 4 per thread
    for (int i = 0; i < 4; ++i) {
      int e = tid + i * 256;
      int row = e >> 3, part = e & 7;
      async_copy_b128(&sA[buf][row * LDA + part * 8],
                      Asrc + (size_t)(mBase + row) * K + ks + part * 8);
    }
    // B tile: 64 rows x 64 halves = 512 x 16B; 2 per thread
    for (int i = 0; i < 2; ++i) {
      int e = tid + i * 256;
      int row = e >> 3, part = e & 7;
      async_copy_b128(&sB[buf][row * LDA + part * 8],
                      Wt + (size_t)(nBase + row) * K + ks + part * 8);
    }
  };

  issue_tile(0, 0);
  int nSteps = K / KS;
  for (int step = 0; step < nSteps; ++step) {
    int buf = step & 1;
    bool hasNext = (step + 1) < nSteps;
    if (hasNext) {
      issue_tile(buf ^ 1, (step + 1) * KS);   // overlap next copy with compute
      wait_async_le6();                       // in-order: older 6 (this tile) done
    } else {
      wait_async_le0();
    }
    __syncthreads();                          // all waves' writes to buf visible
    for (int kc = 0; kc < 2; ++kc) {
      v16h a0 = ld_frag_a(&sA[buf][wm * LDA + kc * 32], LDA);
      v16h a1 = ld_frag_a(&sA[buf][(wm + 16) * LDA + kc * 32], LDA);
      v16h b0 = ld_frag_b(&sB[buf][wn * LDA + kc * 32], LDA);
      v16h b1 = ld_frag_b(&sB[buf][(wn + 16) * LDA + kc * 32], LDA);
      acc00 = wmma16(a0, b0, acc00);
      acc01 = wmma16(a0, b1, acc01);
      acc10 = wmma16(a1, b0, acc10);
      acc11 = wmma16(a1, b1, acc11);
    }
    __syncthreads();                          // reads done before buf is rewritten
  }

  int half8 = (lane < 16) ? 0 : 8;
  int nl = lane & 15;
  v8f accs[2][2] = {{acc00, acc01}, {acc10, acc11}};
  for (int i = 0; i < 2; ++i)
    for (int j = 0; j < 2; ++j) {
      int ng = nBase + wn + j * 16 + nl;
      float bv = bias[ng];
      for (int r = 0; r < 8; ++r) {
        int mg = mBase + wm + i * 16 + r + half8;
        float val = accs[i][j][r] + bv;
        if (MODE == 0) {
          int type = ng >> 8, head = (ng >> 5) & 7, dc = ng & 31;
          int win = mg >> log2L, l = mg & ((1 << log2L) - 1);
          reinterpret_cast<_Float16*>(outp)[(size_t)type * QKV_STRIDE +
              ((((size_t)(win * NHEAD + head)) << log2L) + l) * DH + dc] =
              (_Float16)val;
        } else if (MODE == 1) {
          reinterpret_cast<float*>(outp)[(size_t)mg * N + ng] = val;
        } else {
          reinterpret_cast<_Float16*>(outp)[(size_t)mg * N + ng] =
              (_Float16)fmaxf(val, 0.f);
        }
      }
    }
}

// ---------------- windowed attention, one (window, head) per wave-group ------
template <int L, int WPB>
__global__ __launch_bounds__(256) void attn_kernel(
    const _Float16* __restrict__ Q, const _Float16* __restrict__ Km,
    const _Float16* __restrict__ V, const unsigned char* __restrict__ REG,
    _Float16* __restrict__ O) {
  constexpr int LP = L + 8;            // padded row (mult of 8 halves)
  constexpr int WPP = L / 16;          // waves per (window,head)
  __shared__ __align__(16) _Float16 sVT[WPB][DH][LP];   // V transposed [d][key]
  __shared__ __align__(16) _Float16 sAtt[8][16][LP];    // probs, per wave
  int tid = threadIdx.x, wave = tid >> 5, lane = tid & 31;
  int pairLocal = wave / WPP;
  int wip = wave % WPP;
  int p = blockIdx.x * WPB + pairLocal;
  int win = p >> 3, head = p & 7;
  size_t rowBase = (size_t)(win * NHEAD + head) * L;
  {  // stage V^T cooperatively
    constexpr int TPP = 256 / WPB;
    int lt = tid % TPP, pl2 = tid / TPP;
    int p2 = blockIdx.x * WPB + pl2;
    size_t rb2 = (size_t)((p2 >> 3) * NHEAD + (p2 & 7)) * L;
    for (int i = 0; i < (L * DH) / TPP; ++i) {
      int e = i * TPP + lt;
      int key = e >> 5, d = e & 31;
      sVT[pl2][d][key] = V[(rb2 + key) * DH + d];
    }
  }
  __syncthreads();
  int qrow0 = wip * 16;
  v16h qf = ld_frag_a(Q + (rowBase + qrow0) * DH, DH);  // straight from global
  v8f zero = {0.f, 0.f, 0.f, 0.f, 0.f, 0.f, 0.f, 0.f};
  v8f sc[L / 16];
  for (int nt = 0; nt < L / 16; ++nt) {
    if (nt + 1 < L / 16)
      __builtin_prefetch(Km + (rowBase + (nt + 1) * 16) * DH, 0, 1);
    v16h kf = ld_frag_b(Km + (rowBase + nt * 16) * DH, DH);  // B = K^T
    sc[nt] = wmma16(qf, kf, zero);                            // K-dim = dh = 32
  }
  int half8 = (lane < 16) ? 0 : 8;
  int nl = lane & 15;
  int regBase = win * L;
  unsigned char rq[8];
  for (int r = 0; r < 8; ++r) rq[r] = REG[regBase + qrow0 + r + half8];
  const float scale = 0.17677669529663687f;  // 1/sqrt(32)
  float mx[8], sm[8];
  for (int r = 0; r < 8; ++r) { mx[r] = -3.0e38f; sm[r] = 0.f; }
  for (int nt = 0; nt < L / 16; ++nt) {
    unsigned char rk = REG[regBase + nt * 16 + nl];
    bool padk = (rk & 0x80) != 0;
    for (int r = 0; r < 8; ++r) {
      float s = sc[nt][r] * scale;
      if (padk || ((rk ^ rq[r]) & 0x7f)) s = -1.0e30f;   // shift-window mask
      sc[nt][r] = s;
      mx[r] = fmaxf(mx[r], s);
    }
  }
  for (int off = 1; off < 16; off <<= 1)
    for (int r = 0; r < 8; ++r) mx[r] = fmaxf(mx[r], __shfl_xor(mx[r], off, 32));
  for (int nt = 0; nt < L / 16; ++nt)
    for (int r = 0; r < 8; ++r) {
      float e = __expf(sc[nt][r] - mx[r]);
      sc[nt][r] = e; sm[r] += e;
    }
  for (int off = 1; off < 16; off <<= 1)
    for (int r = 0; r < 8; ++r) sm[r] += __shfl_xor(sm[r], off, 32);
  float inv[8];
  for (int r = 0; r < 8; ++r) inv[r] = 1.f / (sm[r] + 1e-20f);
  for (int nt = 0; nt < L / 16; ++nt)
    for (int r = 0; r < 8; ++r)
      sAtt[wave][r + half8][nt * 16 + nl] = (_Float16)(sc[nt][r] * inv[r]);
  __syncthreads();
  v8f o0 = zero, o1 = zero;
  for (int kc = 0; kc < L / 32; ++kc) {
    v16h af = ld_frag_a(&sAtt[wave][0][kc * 32], LP);
    v16h b0 = ld_frag_b(&sVT[pairLocal][0][kc * 32], LP);
    v16h b1 = ld_frag_b(&sVT[pairLocal][16][kc * 32], LP);
    o0 = wmma16(af, b0, o0);
    o1 = wmma16(af, b1, o1);
  }
  for (int r = 0; r < 8; ++r) {
    int m = qrow0 + r + half8;
    size_t t = (size_t)win * L + m;
    O[t * D_MODEL + head * DH + nl]      = (_Float16)o0[r];
    O[t * D_MODEL + head * DH + 16 + nl] = (_Float16)o1[r];
  }
}

// ---------------- residual + LayerNorm (mode 1 also scatters back to NCHW) ---
__global__ __launch_bounds__(256) void add_ln_kernel(
    const _Float16* __restrict__ Xres16, const float* __restrict__ Xres32,
    const float* __restrict__ Y, const float* __restrict__ g,
    const float* __restrict__ bta, _Float16* __restrict__ out16,
    float* __restrict__ out32, float* __restrict__ scat, LayerP lp, int mode) {
  int wave = threadIdx.x >> 5, lane = threadIdx.x & 31;
  size_t t = (size_t)blockIdx.x * 8 + wave;
  float vals[8];
  float smv = 0.f;
  for (int i = 0; i < 8; ++i) {
    int c = lane + i * 32;
    float x = (mode == 0) ? (float)Xres16[t * D_MODEL + c] : Xres32[t * D_MODEL + c];
    float zz = x + Y[t * D_MODEL + c];
    vals[i] = zz; smv += zz;
  }
  for (int off = 1; off < 32; off <<= 1) smv += __shfl_xor(smv, off, 32);
  float mean = smv * (1.f / 256.f);
  float sq = 0.f;
  for (int i = 0; i < 8; ++i) { float d = vals[i] - mean; sq += d * d; }
  for (int off = 1; off < 32; off <<= 1) sq += __shfl_xor(sq, off, 32);
  float rstd = rsqrtf(sq * (1.f / 256.f) + 1e-5f);
  if (mode == 0) {
    for (int i = 0; i < 8; ++i) {
      int c = lane + i * 32;
      float o = (vals[i] - mean) * rstd * g[c] + bta[c];
      out16[t * D_MODEL + c] = (_Float16)o;
      out32[t * D_MODEL + c] = o;
    }
  } else {  // inverse window_partition + dilate_unfold + roll, write NCHW
    int gw = (int)(t >> lp.log2L);
    int l  = (int)t & (lp.L - 1);
    int beff = gw / lp.NW, nw = gw % lp.NW;
    int b = beff / (lp.s * lp.s);
    int rem = beff % (lp.s * lp.s);
    int si = rem / lp.s, sj = rem % lp.s;
    int wi = nw / lp.nwW, wj = nw % lp.nwW;
    int ph = l / lp.ww, pw = l % lp.ww;
    int hh = wi * lp.wh + ph, wf = wj * lp.ww + pw;
    int hr = hh * lp.s + si, wr = wf * lp.s + sj;
    int hd = (hr + lp.shH) & (HH_ - 1);
    int wd = (wr + lp.shW) & (WW_ - 1);
    for (int i = 0; i < 8; ++i) {
      int c = lane + i * 32;
      float o = (vals[i] - mean) * rstd * g[c] + bta[c];
      scat[(((size_t)b * D_MODEL + c) * HH_ + hd) * WW_ + wd] = o;
    }
  }
}

// ---------------- host orchestration ----------------------------------------
extern "C" void kernel_launch(void* const* d_in, const int* in_sizes, int n_in,
                              void* d_out, int out_size, void* d_ws, size_t ws_size,
                              hipStream_t stream) {
  const float* src   = (const float*)d_in[0];
  const float* pos   = (const float*)d_in[1];
  const unsigned char* mask = (const unsigned char*)d_in[2];
  const float* in_w  = (const float*)d_in[3];
  const float* in_b  = (const float*)d_in[4];
  const float* out_w = (const float*)d_in[5];
  const float* out_b = (const float*)d_in[6];
  const float* l1_w  = (const float*)d_in[7];
  const float* l1_b  = (const float*)d_in[8];
  const float* l2_w  = (const float*)d_in[9];
  const float* l2_b  = (const float*)d_in[10];
  const float* ln1_g = (const float*)d_in[11];
  const float* ln1_b = (const float*)d_in[12];
  const float* ln2_g = (const float*)d_in[13];
  const float* ln2_b = (const float*)d_in[14];

  char* ws = (char*)d_ws;
  size_t off = 0;
  auto take = [&](size_t bytes) {
    char* ptr = ws + off;
    off += (bytes + 255) & ~(size_t)255;
    return ptr;
  };
  _Float16* wQKV = (_Float16*)take((size_t)768 * 256 * 2);
  _Float16* wOut = (_Float16*)take((size_t)256 * 256 * 2);
  _Float16* wL1  = (_Float16*)take((size_t)512 * 256 * 2);
  _Float16* wL2  = (_Float16*)take((size_t)256 * 512 * 2);
  float*    M0   = (float*)take((size_t)T_TOK * D_MODEL * 4);      // evolving memory
  _Float16* X16  = (_Float16*)take((size_t)T_TOK * D_MODEL * 2);
  _Float16* XP16 = (_Float16*)take((size_t)T_TOK * D_MODEL * 2);
  unsigned char* REG = (unsigned char*)take(T_TOK);
  char* regionA = take((size_t)T_TOK * D_MODEL * 2 * 4);           // 128MB reused
  float* Y2 = (float*)take((size_t)T_TOK * D_MODEL * 4);

  _Float16* Qb = (_Float16*)regionA;
  _Float16* Kb = Qb + QKV_STRIDE;
  _Float16* Vb = Kb + QKV_STRIDE;
  float* Y32   = (float*)regionA;                                   // over Q|K (dead)
  float* X1_32 = (float*)(regionA + (size_t)T_TOK * D_MODEL * 4);   // over V|spare
  _Float16* F16 = (_Float16*)regionA;                               // ffn1 out (dead Y)
  _Float16* O16 = XP16;                                             // attn out (dead XP)

  cvt_kernel<<<(768 * 256 + 255) / 256, 256, 0, stream>>>(in_w,  wQKV, 768 * 256);
  cvt_kernel<<<(256 * 256 + 255) / 256, 256, 0, stream>>>(out_w, wOut, 256 * 256);
  cvt_kernel<<<(512 * 256 + 255) / 256, 256, 0, stream>>>(l1_w,  wL1,  512 * 256);
  cvt_kernel<<<(256 * 512 + 255) / 256, 256, 0, stream>>>(l2_w,  wL2,  256 * 512);

  LayerP lps[4];
  const int ss[4] = {1, 1, 2, 2}, whs[4] = {8, 8, 4, 4}, wws[4] = {16, 16, 8, 8};
  for (int i = 0; i < 4; ++i) {
    LayerP& lp = lps[i];
    lp.s = ss[i]; lp.wh = whs[i]; lp.ww = wws[i];
    lp.useMask = i & 1;
    lp.shH = lp.useMask ? lp.wh / 2 : 0;
    lp.shW = lp.useMask ? lp.ww / 2 : 0;
    lp.L = lp.wh * lp.ww;
    lp.log2L = (lp.L == 128) ? 7 : 5;
    lp.newH = HH_ / lp.s; lp.newW = WW_ / lp.s;
    lp.nwW = lp.newW / lp.ww;
    lp.NW = (lp.newH / lp.wh) * lp.nwW;
  }

  for (int i = 0; i < 4; ++i) {
    const LayerP lp = lps[i];
    const float* memIn = (i == 0) ? src : M0;
    float* memOut = (i == 3) ? (float*)d_out : M0;
    gather_kernel<<<(T_TOK * D_MODEL) / 256, 256, 0, stream>>>(
        memIn, pos, mask, X16, XP16, REG, lp);
    gemm_kernel<0><<<dim3(512, 12), 256, 0, stream>>>(
        XP16, X16, wQKV, in_b, (void*)Qb, 256, 768, lp.log2L);
    int pairs = (T_TOK >> lp.log2L) * NHEAD;
    if (lp.L == 128)
      attn_kernel<128, 1><<<pairs, 256, 0, stream>>>(Qb, Kb, Vb, REG, O16);
    else
      attn_kernel<32, 4><<<pairs / 4, 256, 0, stream>>>(Qb, Kb, Vb, REG, O16);
    gemm_kernel<1><<<dim3(512, 4), 256, 0, stream>>>(
        O16, (const _Float16*)nullptr, wOut, out_b, (void*)Y32, 256, 256, lp.log2L);
    add_ln_kernel<<<T_TOK / 8, 256, 0, stream>>>(
        X16, (const float*)nullptr, Y32, ln1_g, ln1_b, X16, X1_32,
        (float*)nullptr, lp, 0);
    gemm_kernel<2><<<dim3(512, 8), 256, 0, stream>>>(
        X16, (const _Float16*)nullptr, wL1, l1_b, (void*)F16, 256, 512, lp.log2L);
    gemm_kernel<1><<<dim3(512, 4), 256, 0, stream>>>(
        F16, (const _Float16*)nullptr, wL2, l2_b, (void*)Y2, 512, 256, lp.log2L);
    add_ln_kernel<<<T_TOK / 8, 256, 0, stream>>>(
        (const _Float16*)nullptr, X1_32, Y2, ln2_g, ln2_b,
        (_Float16*)nullptr, (float*)nullptr, memOut, lp, 1);
  }
}
